// AtomEncoder_13322988552653
// MI455X (gfx1250) — matrix operand order repaired and checked by reading, not compiled, not run on hardware
//
#include <hip/hip_runtime.h>
#include <hip/hip_bf16.h>

// MPNN (NNConv + GRU) for MI455X / gfx1250, wave32 + WMMA.
//
// Per edge-tile of 16: X[16,1056] built on the fly (X[e, k*32+i] =
// r[e,k]*h_src[e,i]; last 32 K-rows carry h_src to fold in b_e2), multiplied
// by constant M[1056,32] (reshaped W_e2 + bias rows) via
// v_wmma_f32_16x16x32_f16. M's B-fragments are pre-permuted into per-lane
// WMMA layout and staged in LDS. W_edge (1.31 GB) is never materialized.

typedef __attribute__((ext_vector_type(16))) _Float16 v16h;
typedef __attribute__((ext_vector_type(8)))  float    v8f;

#define H 32
#define KSTEPS 33                       // 32 r-weighted steps + 1 bias step
#define MFRAG_HALVES (KSTEPS * 2 * 32 * 16)   // 33792 f16 = 67584 B
#define LDS_HALVES   (32 * 2 * 32 * 16)       // first 32 steps -> 65536 B LDS

// ---------------------------------------------------------------- utilities
__global__ void zero_kernel(float* __restrict__ p, int n) {
  int t = blockIdx.x * blockDim.x + threadIdx.x;
  if (t < n) p[t] = 0.0f;
}

__global__ void deg_kernel(const int* __restrict__ dst, float* __restrict__ deg, int E) {
  int t = blockIdx.x * blockDim.x + threadIdx.x;
  if (t < E) atomicAdd(&deg[dst[t]], 1.0f);
}

__global__ void inv_deg_kernel(float* __restrict__ d, int N) {
  int t = blockIdx.x * blockDim.x + threadIdx.x;
  if (t < N) d[t] = 1.0f / fmaxf(d[t], 1.0f);
}

// ------------------------------------------------- h = n_feat @ W_atom + b
__global__ void embed_atoms_kernel(const float* __restrict__ nf, const float* __restrict__ W,
                                   const float* __restrict__ b, float* __restrict__ h, int N) {
  int t = blockIdx.x * blockDim.x + threadIdx.x;
  int n = t >> 5, o = t & 31;
  if (n >= N) return;
  float acc = b[o];
  const float* x = nf + (size_t)n * 64;
#pragma unroll
  for (int k = 0; k < 64; ++k) acc += x[k] * W[k * H + o];
  h[(size_t)n * H + o] = acc;
}

// --------- r = relu( (e_feat@W_bond + b_bond) @ W_e1 + b_e1 )  (wave/edge)
__global__ void bond_edge_kernel(const float* __restrict__ ef, const float* __restrict__ Wb,
                                 const float* __restrict__ bb, const float* __restrict__ We1,
                                 const float* __restrict__ be1, float* __restrict__ rout, int E) {
  int t = blockIdx.x * blockDim.x + threadIdx.x;
  int e = t >> 5, o = t & 31;
  if (e >= E) return;
  float hb = bb[o];
  const float* x = ef + (size_t)e * 16;
#pragma unroll
  for (int k = 0; k < 16; ++k) hb += x[k] * Wb[k * H + o];
  float acc = be1[o];
#pragma unroll
  for (int k = 0; k < H; ++k) acc += __shfl(hb, k, 32) * We1[k * H + o];
  rout[(size_t)e * H + o] = fmaxf(acc, 0.0f);
}

// ------------- pre-permute M[1056,32] into per-lane WMMA B-fragments (f16)
// Layout: [s][nt][lane][j]; col=lane&15, K = 8*(lane>=16) + j + 8*(j>=8).
// M[kk*32+ii, o] = W_e2[kk*1024 + ii*32 + o]; bias rows M[1024+i, o] = b_e2[i*32+o].
__global__ void build_mfrag_kernel(const float* __restrict__ W_e2, const float* __restrict__ b_e2,
                                   _Float16* __restrict__ Mfrag) {
  int t = blockIdx.x * blockDim.x + threadIdx.x;
  if (t >= MFRAG_HALVES) return;
  int j    = t & 15;
  int lane = (t >> 4) & 31;
  int nt   = (t >> 9) & 1;
  int s    = t >> 10;
  int o    = nt * 16 + (lane & 15);
  int kloc = ((lane < 16) ? 0 : 8) + j + ((j >= 8) ? 8 : 0);
  float v  = (s < 32) ? W_e2[(size_t)s * 1024 + kloc * H + o] : b_e2[kloc * H + o];
  Mfrag[t] = (_Float16)v;
}

// -------------------------------------------------- fused NNConv message GEMM
__global__ __launch_bounds__(256) void msg_wmma_kernel(
    const float* __restrict__ h,        // [N,H] current node states
    const float* __restrict__ rbuf,     // [E,H] relu stage-1 per edge
    const _Float16* __restrict__ Mfrag, // [33][2][32][16] pre-permuted B frags
    const int* __restrict__ src, const int* __restrict__ dst,
    float* __restrict__ agg,            // [N,H], pre-zeroed; raw sums (scaled later)
    int nEdges) {
  __shared__ _Float16 lM[LDS_HALVES];   // 64 KB: B fragments for s = 0..31
  {
    const uint32_t* g32 = (const uint32_t*)Mfrag;
    uint32_t* l32 = (uint32_t*)lM;
    for (int i = threadIdx.x; i < LDS_HALVES / 2; i += blockDim.x) l32[i] = g32[i];
  }
  __syncthreads();

  const int lane = threadIdx.x & 31;
  const int wavesPerGrid = (gridDim.x * blockDim.x) >> 5;
  const int waveId = (blockIdx.x * blockDim.x + threadIdx.x) >> 5;
  const int nTiles = (nEdges + 15) >> 4;

  const int m    = lane & 15;               // edge row handled by this lane
  const int base = (lane < 16) ? 0 : 8;     // A/C half-select (ISA 16-bit layout)
  const int col  = lane & 15;               // output column within 16-wide tile

  for (int tile = waveId; tile < nTiles; tile += wavesPerGrid) {
    int e  = tile * 16 + m;
    int ec = (e < nEdges) ? e : (nEdges - 1);   // clamp; dup rows never scattered
    int sn = src[ec];

    // h[src] samples in this lane's A K-pattern: K = base + j + 8*(j>=8)
    const float* hp = h + (size_t)sn * H;
    _Float16 hvh[16];
#pragma unroll
    for (int j = 0; j < 8; ++j) hvh[j]     = (_Float16)hp[base + j];
#pragma unroll
    for (int j = 0; j < 8; ++j) hvh[8 + j] = (_Float16)hp[base + 16 + j];

    const float* rp = rbuf + (size_t)ec * H;

    v8f acc0 = {}; v8f acc1 = {};
    // Outer K-loop NOT unrolled: keeps ds_load addresses loop-variant so the
    // 64 LDS B-fragments are never hoisted out of the tile loop (no spills).
    // Inner 4 steps fully unrolled: only constant VGPR indices (no movrel).
#pragma unroll 1
    for (int s4 = 0; s4 < 32; s4 += 4) {
      const float4 rq = *(const float4*)(rp + s4);
      _Float16 r4[4];
      r4[0] = (_Float16)rq.x; r4[1] = (_Float16)rq.y;
      r4[2] = (_Float16)rq.z; r4[3] = (_Float16)rq.w;
#pragma unroll
      for (int u = 0; u < 4; ++u) {
        _Float16 rs = r4[u];
        v16h a;
#pragma unroll
        for (int j = 0; j < 16; ++j) a[j] = rs * hvh[j];   // X rows: r_s * h_src
        int s = s4 + u;
        v16h b0 = *(const v16h*)&lM[((s * 2 + 0) * 32 + lane) * 16];
        v16h b1 = *(const v16h*)&lM[((s * 2 + 1) * 32 + lane) * 16];
        acc0 = __builtin_amdgcn_wmma_f32_16x16x32_f16(false, a, false, b0, (short)0, acc0, false, false);
        acc1 = __builtin_amdgcn_wmma_f32_16x16x32_f16(false, a, false, b1, (short)0, acc1, false, false);
      }
    }
    {   // bias K-step (rows 1024..1055 of M): A row is h_src itself
      v16h a;
#pragma unroll
      for (int j = 0; j < 16; ++j) a[j] = hvh[j];
      v16h b0 = *(const v16h*)&Mfrag[((32 * 2 + 0) * 32 + lane) * 16];
      v16h b1 = *(const v16h*)&Mfrag[((32 * 2 + 1) * 32 + lane) * 16];
      acc0 = __builtin_amdgcn_wmma_f32_16x16x32_f16(false, a, false, b0, (short)0, acc0, false, false);
      acc1 = __builtin_amdgcn_wmma_f32_16x16x32_f16(false, a, false, b1, (short)0, acc1, false, false);
    }
    // scatter: C/D layout -> lane holds (M = v + base, N = col)
#pragma unroll
    for (int v = 0; v < 8; ++v) {
      int ee = tile * 16 + v + base;
      if (ee < nEdges) {
        int d = dst[ee];
        atomicAdd(&agg[(size_t)d * H + col],      acc0[v]);
        atomicAdd(&agg[(size_t)d * H + 16 + col], acc1[v]);
      }
    }
  }
}

// ------------------------------------------------------------- GRU update
__global__ void gru_kernel(const float* __restrict__ agg, const float* __restrict__ invd,
                           const float* __restrict__ h_in, float* __restrict__ h_out,
                           const float* __restrict__ W_ih, const float* __restrict__ W_hh,
                           const float* __restrict__ b_ih, const float* __restrict__ b_hh, int N) {
  int t = blockIdx.x * blockDim.x + threadIdx.x;
  int n = t >> 5, o = t & 31;
  if (n >= N) return;
  float mval = fmaxf(agg[(size_t)n * H + o] * invd[n], 0.0f);   // mean + relu
  float hval = h_in[(size_t)n * H + o];
  float gir = b_ih[o], giz = b_ih[H + o], gig = b_ih[2 * H + o];
  float ghr = b_hh[o], ghz = b_hh[H + o], ghg = b_hh[2 * H + o];
#pragma unroll
  for (int k = 0; k < H; ++k) {
    float mk = __shfl(mval, k, 32);
    float hk = __shfl(hval, k, 32);
    gir += mk * W_ih[o * H + k];
    giz += mk * W_ih[(H + o) * H + k];
    gig += mk * W_ih[(2 * H + o) * H + k];
    ghr += hk * W_hh[o * H + k];
    ghz += hk * W_hh[(H + o) * H + k];
    ghg += hk * W_hh[(2 * H + o) * H + k];
  }
  float rr = 1.0f / (1.0f + __expf(-(gir + ghr)));
  float zz = 1.0f / (1.0f + __expf(-(giz + ghz)));
  float nn = tanhf(gig + rr * ghg);
  h_out[(size_t)n * H + o] = (1.0f - zz) * nn + zz * hval;
}

// ------------------------------------------------------------- readout
__global__ void readout_accum_kernel(const float* __restrict__ h, const int* __restrict__ gid,
                                     float* __restrict__ gsum, float* __restrict__ gcnt, int N) {
  int t = blockIdx.x * blockDim.x + threadIdx.x;
  int n = t >> 5, o = t & 31;
  if (n >= N) return;
  int g = gid[n];
  atomicAdd(&gsum[(size_t)g * H + o], h[(size_t)n * H + o]);
  if (o == 0) atomicAdd(&gcnt[g], 1.0f);
}

__global__ void readout_final_kernel(const float* __restrict__ gsum, const float* __restrict__ gcnt,
                                     float* __restrict__ out, int total) {
  int t = blockIdx.x * blockDim.x + threadIdx.x;
  if (t >= total) return;
  out[t] = gsum[t] / fmaxf(gcnt[t >> 5], 1.0f);
}

// ------------------------------------------------------------- launcher
extern "C" void kernel_launch(void* const* d_in, const int* in_sizes, int n_in,
                              void* d_out, int out_size, void* d_ws, size_t ws_size,
                              hipStream_t stream) {
  (void)n_in; (void)ws_size;
  const float* n_feat = (const float*)d_in[0];
  const float* e_feat = (const float*)d_in[1];
  const int*   src    = (const int*)d_in[2];
  const int*   dst    = (const int*)d_in[3];
  const int*   gid    = (const int*)d_in[4];
  const float* W_atom = (const float*)d_in[5];
  const float* b_atom = (const float*)d_in[6];
  const float* W_bond = (const float*)d_in[7];
  const float* b_bond = (const float*)d_in[8];
  const float* W_e1   = (const float*)d_in[9];
  const float* b_e1   = (const float*)d_in[10];
  const float* W_e2   = (const float*)d_in[11];
  const float* b_e2   = (const float*)d_in[12];
  const float* W_ih   = (const float*)d_in[13];
  const float* W_hh   = (const float*)d_in[14];
  const float* b_ih   = (const float*)d_in[15];
  const float* b_hh   = (const float*)d_in[16];
  float* out = (float*)d_out;

  const int N = in_sizes[0] / 64;
  const int E = in_sizes[1] / 16;
  const int B = out_size / H;

  // workspace layout (fp32 units); ~49 MB total
  float* ws = (float*)d_ws;
  size_t off = 0;
  float* hA   = ws + off; off += (size_t)N * H;
  float* hB   = ws + off; off += (size_t)N * H;
  float* agg  = ws + off; off += (size_t)N * H;
  float* invd = ws + off; off += (size_t)N;
  float* rbuf = ws + off; off += (size_t)E * H;
  float* gsum = ws + off; off += (size_t)B * H;
  float* gcnt = ws + off; off += (size_t)B;
  off = (off + 15) & ~(size_t)15;                 // 64B-align Mfrag for v16h loads
  _Float16* Mfrag = (_Float16*)(ws + off);

  auto cdiv = [](int a, int b) { return (a + b - 1) / b; };

  // degrees -> inverse degrees (in place)
  zero_kernel<<<cdiv(N, 256), 256, 0, stream>>>(invd, N);
  deg_kernel<<<cdiv(E, 256), 256, 0, stream>>>(dst, invd, E);
  inv_deg_kernel<<<cdiv(N, 256), 256, 0, stream>>>(invd, N);

  embed_atoms_kernel<<<cdiv(N * H, 256), 256, 0, stream>>>(n_feat, W_atom, b_atom, hA, N);
  bond_edge_kernel<<<cdiv(E * H, 256), 256, 0, stream>>>(e_feat, W_bond, b_bond, W_e1, b_e1, rbuf, E);
  build_mfrag_kernel<<<cdiv(MFRAG_HALVES, 256), 256, 0, stream>>>(W_e2, b_e2, Mfrag);

  const float* hin = hA;
  float* hout = hB;
  for (int layer = 0; layer < 2; ++layer) {
    zero_kernel<<<cdiv(N * H, 256), 256, 0, stream>>>(agg, N * H);
    msg_wmma_kernel<<<1024, 256, 0, stream>>>(hin, rbuf, Mfrag, src, dst, agg, E);
    gru_kernel<<<cdiv(N * H, 256), 256, 0, stream>>>(agg, invd, hin, hout, W_ih, W_hh, b_ih, b_hh, N);
    const float* tmp = hout; hout = (float*)hin; hin = tmp;
  }
  // after 2 layers, hin (== hA) holds the final node states
  zero_kernel<<<cdiv(B * H, 256), 256, 0, stream>>>(gsum, B * H);
  zero_kernel<<<cdiv(B, 256), 256, 0, stream>>>(gcnt, B);
  readout_accum_kernel<<<cdiv(N * H, 256), 256, 0, stream>>>(hin, gid, gsum, gcnt, N);
  readout_final_kernel<<<cdiv(out_size, 256), 256, 0, stream>>>(gsum, gcnt, out, out_size);
}
